// PtConv_57372173140529
// MI455X (gfx1250) — matrix operand order repaired and checked by reading, not compiled, not run on hardware
//
#include <hip/hip_runtime.h>

#define Bc   8
#define Nc   8192
#define CINc 64
#define COUTc 64
#define Kc   16
#define KNc  16

typedef __attribute__((ext_vector_type(16))) __bf16 v16bf;
typedef __attribute__((ext_vector_type(8)))  float  v8f;

struct Q2 { uint4 a, b; };
static __device__ __forceinline__ v16bf frag(uint4 lo, uint4 hi) {
  Q2 q{lo, hi};
  return __builtin_bit_cast(v16bf, q);
}

// ---------------------------------------------------------------------------
// Prep kernel: pack W -> bf16 column-major with K reindexed q = j*64 + c,
// and fold layer-1 (rank-3 trick): A1[dim][o] = sum_j l1_w[(dim,j),o],
// c1[o] = l1_b[o] - sum_{dim,j} centers[dim,j] * l1_w[(dim,j),o].
// ---------------------------------------------------------------------------
__global__ __launch_bounds__(256) void ptconv_prep(
    const float* __restrict__ weight,    // [CIN][KN][COUT]
    const float* __restrict__ l1w,       // [48][32]
    const float* __restrict__ l1b,       // [32]
    const float* __restrict__ centers,   // [3][16]
    __bf16* __restrict__ W2,             // [COUT][1024], q = j*64 + c
    float* __restrict__ A1,              // [3][32]
    float* __restrict__ c1)              // [32]
{
  int t = blockIdx.x * 256 + threadIdx.x;
  if (t < COUTc * 1024) {
    int o = t >> 10, q = t & 1023;
    int j = q >> 6, c = q & 63;
    W2[t] = (__bf16)weight[(c * KNc + j) * COUTc + o];
  }
  if (blockIdx.x == 0 && threadIdx.x < 32) {
    int o = threadIdx.x;
    float cc = l1b[o];
    for (int dim = 0; dim < 3; ++dim) {
      float s = 0.f;
      for (int jn = 0; jn < 16; ++jn) {
        float w = l1w[(dim * 16 + jn) * 32 + o];
        s += w;
        cc -= centers[dim * 16 + jn] * w;
      }
      A1[dim * 32 + o] = s;
    }
    c1[o] = cc;
  }
}

// Dynamic LDS layout ---------------------------------------------------------
#define SMALL_BYTES 3712              // 928 floats of staged small weights
#define OFF_HT   0                    // [16 pts][16 j][16 k] bf16  (8 KB)
#define OFF_GFT  8192                 // [64 c][16 k] bf16          (2 KB)
#define OFF_AGG  10240                // [16 pts][1024 q] bf16      (32 KB)
#define OFF_IDX  43008                // [256] int                  (1 KB)
#define WAVE_BYTES 44032
// 2 waves per block -> 91,776 B: 3 blocks (6 waves) fit per 320 KB WGP.
#define SMEM_BYTES (SMALL_BYTES + 2 * WAVE_BYTES)

// ---------------------------------------------------------------------------
// Main kernel: 1 wave per 16-point tile, 2 waves per block.
// ---------------------------------------------------------------------------
__global__ __launch_bounds__(64) void ptconv_main(
    const float* __restrict__ features,
    const float* __restrict__ input_pts,
    const float* __restrict__ output_pts,
    const float* __restrict__ bias,
    const float* __restrict__ l2w, const float* __restrict__ l2b,
    const float* __restrict__ l3w, const float* __restrict__ l3b,
    const int*   __restrict__ indices,
    const __bf16* __restrict__ W2,
    const float* __restrict__ A1g, const float* __restrict__ c1g,
    float* __restrict__ out)
{
  extern __shared__ char smem[];
  float* s_A1  = (float*)smem;          // 96
  float* s_c1  = s_A1 + 96;             // 32
  float* s_l2w = s_A1 + 128;            // 512
  float* s_l3w = s_A1 + 640;            // 256
  float* s_l2b = s_A1 + 896;            // 16
  float* s_l3b = s_A1 + 912;            // 16

  const int tid  = threadIdx.x;
  const int lane = tid & 31;
  const int wave = tid >> 5;
  char* wbase = smem + SMALL_BYTES + wave * WAVE_BYTES;
  __bf16* hT  = (__bf16*)(wbase + OFF_HT);
  __bf16* gfT = (__bf16*)(wbase + OFF_GFT);
  __bf16* agg = (__bf16*)(wbase + OFF_AGG);
  int*    sidx = (int*)(wbase + OFF_IDX);

  // Stage small weights (block-cooperative), one barrier for whole kernel.
  for (int i = tid; i < 96;  i += 64) s_A1[i]  = A1g[i];
  if (tid < 32) s_c1[tid] = c1g[tid];
  for (int i = tid; i < 512; i += 64) s_l2w[i] = l2w[i];
  for (int i = tid; i < 256; i += 64) s_l3w[i] = l3w[i];
  if (tid < 16) { s_l2b[tid] = l2b[tid]; s_l3b[tid] = l3b[tid]; }
  __syncthreads();

  const int tile = blockIdx.x * 2 + wave;          // 4096 tiles total
  const int b    = tile >> 9;                      // N/16 = 512 tiles/batch
  const int m0   = (tile & 511) << 4;
  const size_t baseBN = (size_t)b * Nc;

  // ---------------- Phase 1: MLP per (m,k) row -> hT (transposed, bf16) ----
  // Batch all index + point loads first (one wait), then compute.
  int   idxs[8];
  float rr[8][3];
  #pragma unroll
  for (int i = 0; i < 8; ++i) {
    const int r = i * 32 + lane;                   // row = m*16 + k
    const int m = r >> 4, k = r & 15;
    idxs[i] = indices[(baseBN + m0 + m) * Kc + k];
  }
  #pragma unroll
  for (int i = 0; i < 8; ++i) {
    const int r = i * 32 + lane;
    const int m = r >> 4;
    sidx[r] = idxs[i];
    const float* gp = input_pts  + (baseBN + idxs[i]) * 3;
    const float* op = output_pts + (baseBN + m0 + m) * 3;
    rr[i][0] = gp[0] - op[0];
    rr[i][1] = gp[1] - op[1];
    rr[i][2] = gp[2] - op[2];
  }
  #pragma unroll 1
  for (int i = 0; i < 8; ++i) {
    const int r = i * 32 + lane;
    const int m = r >> 4, k = r & 15;
    const float r0 = rr[i][0], r1 = rr[i][1], r2 = rr[i][2];

    float z1[32];
    #pragma unroll
    for (int o = 0; o < 32; ++o) {
      float z = fmaf(r0, s_A1[o], fmaf(r1, s_A1[32 + o],
                fmaf(r2, s_A1[64 + o], s_c1[o])));
      z1[o] = fmaxf(z, 0.f);
    }
    float z2[16];
    #pragma unroll
    for (int o = 0; o < 16; ++o) z2[o] = s_l2b[o];
    #pragma unroll 4
    for (int ii = 0; ii < 32; ++ii) {
      #pragma unroll
      for (int o = 0; o < 16; ++o) z2[o] = fmaf(z1[ii], s_l2w[ii * 16 + o], z2[o]);
    }
    float h[16];
    #pragma unroll
    for (int o = 0; o < 16; ++o) h[o] = s_l3b[o];
    #pragma unroll 4
    for (int ii = 0; ii < 16; ++ii) {
      const float z = fmaxf(z2[ii], 0.f);
      #pragma unroll
      for (int o = 0; o < 16; ++o) h[o] = fmaf(z, s_l3w[ii * 16 + o], h[o]);
    }
    #pragma unroll
    for (int j = 0; j < 16; ++j)
      hT[m * 256 + j * 16 + k] = (__bf16)fmaxf(h[j], 0.f);
  }

  // ---------------- Phase 2: gather + agg WMMA per point -------------------
  // Software-pipelined: point m+1's 8 feature loads fly during point m's
  // WMMAs; all 4 B-fragments are loaded before the 4 back-to-back WMMAs so
  // the D->VALU hazard of each WMMA is covered by the next WMMA, not v_nops.
  const uint4 zero4 = {0u, 0u, 0u, 0u};
  const int  myk = (lane >> 4);                 // k parity handled by lane half
  const int  c4  = (lane & 15) * 4;

  float4 fb[8];
  #pragma unroll
  for (int it = 0; it < 8; ++it) {
    const int idxv = sidx[it * 2 + myk];        // point 0
    fb[it] = *(const float4*)(features + (baseBN + idxv) * CINc + c4);
  }

  #pragma unroll 1
  for (int m = 0; m < 16; ++m) {
    // Commit current point's gathered features transposed: gfT[c][k], bf16.
    #pragma unroll
    for (int it = 0; it < 8; ++it) {
      const int k = it * 2 + myk;
      gfT[(c4 + 0) * 16 + k] = (__bf16)fb[it].x;
      gfT[(c4 + 1) * 16 + k] = (__bf16)fb[it].y;
      gfT[(c4 + 2) * 16 + k] = (__bf16)fb[it].z;
      gfT[(c4 + 3) * 16 + k] = (__bf16)fb[it].w;
    }
    // Prefetch next point's features (overlaps with WMMAs below).
    float4 fn[8];
    if (m < 15) {
      #pragma unroll
      for (int it = 0; it < 8; ++it) {
        const int idxv = sidx[(m + 1) * 16 + it * 2 + myk];
        fn[it] = *(const float4*)(features + (baseBN + idxv) * CINc + c4);
      }
    }
    // A = h^T (M=j, K=k, padded 16->32): 8 halfs per lane + zeros.
    const uint4 alo = *(const uint4*)(hT + m * 256 + (lane & 15) * 16 + (lane >> 4) * 8);
    const v16bf a = frag(alo, zero4);

    // Load all 4 B-fragments (elements 8..15 carry K=16..31 padding -> zero).
    uint4 blo[4], bhi[4];
    #pragma unroll
    for (int t = 0; t < 4; ++t) { blo[t] = zero4; bhi[t] = zero4; }
    {
      #pragma unroll
      for (int t = 0; t < 4; ++t) {
        const __bf16* gp = gfT + (t * 16 + (lane & 15)) * 16 + (lane >> 4) * 8;
        blo[t] = *(const uint4*)gp;
      }
    }
    // 4 independent WMMAs back-to-back.
    const v8f dz = {0.f,0.f,0.f,0.f,0.f,0.f,0.f,0.f};
    v8f d0 = __builtin_amdgcn_wmma_f32_16x16x32_bf16(false, a, false,
              frag(blo[0], bhi[0]), (short)0, dz, false, false);
    v8f d1 = __builtin_amdgcn_wmma_f32_16x16x32_bf16(false, a, false,
              frag(blo[1], bhi[1]), (short)0, dz, false, false);
    v8f d2 = __builtin_amdgcn_wmma_f32_16x16x32_bf16(false, a, false,
              frag(blo[2], bhi[2]), (short)0, dz, false, false);
    v8f d3 = __builtin_amdgcn_wmma_f32_16x16x32_bf16(false, a, false,
              frag(blo[3], bhi[3]), (short)0, dz, false, false);

    // D[j, c-tile] -> agg[m][q = j*64 + c], contiguous across lanes.
    #pragma unroll
    for (int v = 0; v < 8; ++v) {
      const int j = v + (lane >> 4) * 8;
      __bf16* ap = agg + m * 1024 + j * 64 + (lane & 15);
      ap[ 0] = (__bf16)d0[v];
      ap[16] = (__bf16)d1[v];
      ap[32] = (__bf16)d2[v];
      ap[48] = (__bf16)d3[v];
    }
    if (m < 15) {
      #pragma unroll
      for (int it = 0; it < 8; ++it) fb[it] = fn[it];
    }
  }

  // ---------------- Phase 3: out[16,64] = agg[16,1024] @ W ----------------
  v8f acc0 = {0.f,0.f,0.f,0.f,0.f,0.f,0.f,0.f};
  v8f acc1 = acc0, acc2 = acc0, acc3 = acc0;
  #pragma unroll 1
  for (int kt = 0; kt < 32; ++kt) {
    // Batch A-fragment (LDS) + all 4 W-fragments (global, L2-resident).
    const __bf16* ap = agg + (lane & 15) * 1024 + kt * 32 + (lane >> 4) * 8;
    const int kh = kt * 32 + (lane >> 4) * 8;
    uint4 wlo[4], whi[4];
    #pragma unroll
    for (int n = 0; n < 4; ++n) {
      const __bf16* wp = W2 + ((lane & 15) + n * 16) * 1024 + kh;
      wlo[n] = *(const uint4*)wp;
      whi[n] = *(const uint4*)(wp + 16);
    }
    const v16bf a = frag(*(const uint4*)ap, *(const uint4*)(ap + 16));

    acc0 = __builtin_amdgcn_wmma_f32_16x16x32_bf16(false, a, false,
            frag(wlo[0], whi[0]), (short)0, acc0, false, false);
    acc1 = __builtin_amdgcn_wmma_f32_16x16x32_bf16(false, a, false,
            frag(wlo[1], whi[1]), (short)0, acc1, false, false);
    acc2 = __builtin_amdgcn_wmma_f32_16x16x32_bf16(false, a, false,
            frag(wlo[2], whi[2]), (short)0, acc2, false, false);
    acc3 = __builtin_amdgcn_wmma_f32_16x16x32_bf16(false, a, false,
            frag(wlo[3], whi[3]), (short)0, acc3, false, false);
  }

  // ---------------- Epilogue: /K + bias -----------------------------------
  const float inv = 1.0f / 16.0f;
  const float bia0 = bias[(lane & 15)];
  const float bia1 = bias[(lane & 15) + 16];
  const float bia2 = bias[(lane & 15) + 32];
  const float bia3 = bias[(lane & 15) + 48];
  #pragma unroll
  for (int v = 0; v < 8; ++v) {
    const int row = v + (lane >> 4) * 8;
    float* op = out + (baseBN + m0 + row) * COUTc;
    op[(lane & 15)]      = acc0[v] * inv + bia0;
    op[(lane & 15) + 16] = acc1[v] * inv + bia1;
    op[(lane & 15) + 32] = acc2[v] * inv + bia2;
    op[(lane & 15) + 48] = acc3[v] * inv + bia3;
  }
}

// ---------------------------------------------------------------------------
extern "C" void kernel_launch(void* const* d_in, const int* in_sizes, int n_in,
                              void* d_out, int out_size, void* d_ws, size_t ws_size,
                              hipStream_t stream) {
  (void)in_sizes; (void)n_in; (void)out_size; (void)ws_size;
  const float* features   = (const float*)d_in[0];
  const float* input_pts  = (const float*)d_in[1];
  const float* output_pts = (const float*)d_in[2];
  const float* weight     = (const float*)d_in[3];
  const float* bias       = (const float*)d_in[4];
  const float* centers    = (const float*)d_in[5];
  const float* l1_w       = (const float*)d_in[6];
  const float* l1_b       = (const float*)d_in[7];
  const float* l2_w       = (const float*)d_in[8];
  const float* l2_b       = (const float*)d_in[9];
  const float* l3_w       = (const float*)d_in[10];
  const float* l3_b       = (const float*)d_in[11];
  const int*   indices    = (const int*)d_in[12];
  float* out = (float*)d_out;

  __bf16* W2 = (__bf16*)d_ws;                       // 131072 B
  float*  A1 = (float*)((char*)d_ws + 131072);      // 96 floats
  float*  c1 = A1 + 96;                             // 32 floats

  ptconv_prep<<<256, 256, 0, stream>>>(weight, l1_w, l1_b, centers, W2, A1, c1);

  hipFuncSetAttribute((const void*)ptconv_main,
                      hipFuncAttributeMaxDynamicSharedMemorySize, SMEM_BYTES);
  // 4096 tiles of 16 points; 2 waves (tiles) per 64-thread block.
  ptconv_main<<<2048, 64, SMEM_BYTES, stream>>>(
      features, input_pts, output_pts, bias,
      l2_w, l2_b, l3_w, l3_b, indices, W2, A1, c1, out);
}
